// MLAv2_2095944040834
// MI455X (gfx1250) — matrix operand order repaired and checked by reading, not compiled, not run on hardware
//
#include <hip/hip_runtime.h>
#include <hip/hip_bf16.h>

// ---------------------------------------------------------------------------
// MLA forward for MI455X (gfx1250, wave32, WMMA).
// Pipeline: f32->bf16 convert -> GEMMs (bf16 WMMA, f32 accum) -> RoPE ->
//           flash attention (WMMA QK^T + PV, online softmax) -> O-proj (f32 out)
// GEMM: 128x128 block tile, BK=64, double-buffered LDS (register-staged),
//       8 waves x (32x64 register tile) = 16 WMMAs per wave per stage,
//       global_prefetch two stages ahead.
// Attention: 64-key inner tiles, 16 WMMAs per stage per wave.
// ---------------------------------------------------------------------------

typedef __bf16 bf16_t;
typedef bf16_t bf16x8 __attribute__((ext_vector_type(8)));
typedef bf16_t v16bf  __attribute__((ext_vector_type(16)));
typedef float  v8f    __attribute__((ext_vector_type(8)));

__device__ __forceinline__ v16bf ld_frag(const bf16_t* p0, const bf16_t* p1) {
    bf16x8 a = *(const bf16x8*)p0;
    bf16x8 b = *(const bf16x8*)p1;
    v16bf r;
#pragma unroll
    for (int i = 0; i < 8; ++i) { r[i] = a[i]; r[i + 8] = b[i]; }
    return r;
}

__device__ __forceinline__ v8f wmma_bf16(v16bf a, v16bf b, v8f c) {
    // D = A(16x32 bf16) * B(32x16 bf16) + C(16x16 f32)
    return __builtin_amdgcn_wmma_f32_16x16x32_bf16(false, a, false, b,
                                                   (short)0, c, false, false);
}

// ---------------------------------------------------------------------------
// f32 -> bf16 convert
// ---------------------------------------------------------------------------
__global__ __launch_bounds__(256) void f32_to_bf16(const float* __restrict__ in,
                                                   bf16_t* __restrict__ out, int n) {
    int i = blockIdx.x * 256 + threadIdx.x;
    if (i < n) out[i] = (bf16_t)in[i];
}

// ---------------------------------------------------------------------------
// RoPE in-place on bf16 tensor laid out (B*S, H*64). One thread per even/odd
// pair: i = pair index within head (0..31).
// ---------------------------------------------------------------------------
__global__ __launch_bounds__(256) void rope_bf16(bf16_t* __restrict__ X, int total, int S) {
    int idx = blockIdx.x * 256 + threadIdx.x;
    if (idx >= total) return;
    int i   = idx & 31;          // pair index within head (theta index)
    int h   = (idx >> 5) & 15;   // head
    int row = idx >> 9;          // b*S + s
    int pos = row % S;
    size_t base = (size_t)row * 1024 + h * 64 + i * 2;
    float e = (float)X[base];
    float o = (float)X[base + 1];
    float freq = __powf(10000.0f, -(float)(2 * i) / 64.0f);
    float ang  = (float)pos * freq;
    float c = __cosf(ang), sn = __sinf(ang);
    X[base]     = (bf16_t)(e * c - o * sn);
    X[base + 1] = (bf16_t)(e * sn + o * c);
}

// ---------------------------------------------------------------------------
// Tiled bf16 GEMM:  C[M,N] = A[M,K] * B[K,N]   (A,B row-major bf16)
// Block: 256 threads = 8 waves; block tile 128(M) x 128(N); K stage = 64.
// Double-buffered LDS: next stage is loaded into registers while this
// stage's 16 WMMAs execute, then stored into the other LDS buffer.
// ---------------------------------------------------------------------------
template <bool BF16OUT>
__global__ __launch_bounds__(256) void gemm_bf16(const bf16_t* __restrict__ A,
                                                 const bf16_t* __restrict__ B,
                                                 void* __restrict__ Cout,
                                                 int M, int N, int K) {
    __shared__ bf16_t As[2][128][72];   // [buf][row][k], padded stride
    __shared__ bf16_t BsT[2][128][72];  // [buf][col][k], padded stride

    const int tid  = threadIdx.x;
    const int wave = tid >> 5;
    const int lane = tid & 31;
    const int wm   = wave & 3;
    const int wn   = wave >> 2;
    const int m16  = lane & 15;
    const int hi   = lane >> 4;

    const int rowBase = blockIdx.y * 128;
    const int colBase = blockIdx.x * 128;

    // per-thread cooperative-load coordinates (A: 128x64, B: 64x128)
    const int rA[4]   = {(tid + 0) >> 3, (tid + 256) >> 3, (tid + 512) >> 3, (tid + 768) >> 3};
    const int segA    = (tid & 7) * 8;
    const int krB[4]  = {(tid + 0) >> 4, (tid + 256) >> 4, (tid + 512) >> 4, (tid + 768) >> 4};
    const int segB    = (tid & 15) * 8;

    v8f acc[2][4];
#pragma unroll
    for (int s = 0; s < 2; ++s)
#pragma unroll
        for (int t = 0; t < 4; ++t) acc[s][t] = v8f{};

    bf16x8 aReg[4], bReg[4];

    // --- preload stage 0 into registers, then LDS buffer 0 ---
#pragma unroll
    for (int u = 0; u < 4; ++u)
        aReg[u] = *(const bf16x8*)&A[(size_t)(rowBase + rA[u]) * K + segA];
#pragma unroll
    for (int u = 0; u < 4; ++u)
        bReg[u] = *(const bf16x8*)&B[(size_t)krB[u] * N + colBase + segB];
#pragma unroll
    for (int u = 0; u < 4; ++u)
        *(bf16x8*)&As[0][rA[u]][segA] = aReg[u];
#pragma unroll
    for (int u = 0; u < 4; ++u)
#pragma unroll
        for (int i = 0; i < 8; ++i) BsT[0][segB + i][krB[u]] = bReg[u][i];

    int cur = 0;
    for (int k0 = 0; k0 < K; k0 += 64) {
        __syncthreads();  // LDS[cur] fully written by all waves
        const bool more = (k0 + 64) < K;
        if (more) {
            // issue global loads for next stage (latency hidden by WMMAs below)
#pragma unroll
            for (int u = 0; u < 4; ++u)
                aReg[u] = *(const bf16x8*)&A[(size_t)(rowBase + rA[u]) * K + k0 + 64 + segA];
#pragma unroll
            for (int u = 0; u < 4; ++u)
                bReg[u] = *(const bf16x8*)&B[(size_t)(k0 + 64 + krB[u]) * N + colBase + segB];
            if (k0 + 128 < K) {
                __builtin_prefetch(&A[(size_t)(rowBase + rA[0]) * K + k0 + 128 + segA], 0, 1);
                __builtin_prefetch(&B[(size_t)(k0 + 128 + krB[0]) * N + colBase + segB], 0, 1);
            }
        }

        // --- compute on LDS[cur]: 2 K-substeps x 8 WMMAs ---
#pragma unroll
        for (int ks2 = 0; ks2 < 2; ++ks2) {
            v16bf af[2], bfr[4];
#pragma unroll
            for (int s = 0; s < 2; ++s) {
                const int row = wm * 32 + s * 16 + m16;
                // A layout: elems 0..7 -> K = hi*8+i ; elems 8..15 -> K = 16+hi*8+i
                af[s] = ld_frag(&As[cur][row][ks2 * 32 + hi * 8],
                                &As[cur][row][ks2 * 32 + 16 + hi * 8]);
            }
#pragma unroll
            for (int t = 0; t < 4; ++t) {
                const int col = wn * 64 + t * 16 + m16;
                // B layout: elems 0..7 -> K = hi*8+i ; elems 8..15 -> K = 16+hi*8+i
                bfr[t] = ld_frag(&BsT[cur][col][ks2 * 32 + hi * 8],
                                 &BsT[cur][col][ks2 * 32 + 16 + hi * 8]);
            }
#pragma unroll
            for (int s = 0; s < 2; ++s)
#pragma unroll
                for (int t = 0; t < 4; ++t)
                    acc[s][t] = wmma_bf16(af[s], bfr[t], acc[s][t]);
        }

        if (more) {
            // store next stage into the other buffer (no conflict with readers)
            const int nxt = cur ^ 1;
#pragma unroll
            for (int u = 0; u < 4; ++u)
                *(bf16x8*)&As[nxt][rA[u]][segA] = aReg[u];
#pragma unroll
            for (int u = 0; u < 4; ++u)
#pragma unroll
                for (int i = 0; i < 8; ++i) BsT[nxt][segB + i][krB[u]] = bReg[u][i];
        }
        cur ^= 1;
    }

    // --- epilogue: C elem j -> row = j + 8*hi, col = lane%16 ---
#pragma unroll
    for (int s = 0; s < 2; ++s) {
#pragma unroll
        for (int j = 0; j < 8; ++j) {
            const int r = rowBase + wm * 32 + s * 16 + j + 8 * hi;
#pragma unroll
            for (int t = 0; t < 4; ++t) {
                const int c = colBase + wn * 64 + t * 16 + m16;
                if (BF16OUT) {
                    ((bf16_t*)Cout)[(size_t)r * N + c] = (bf16_t)acc[s][t][j];
                } else {
                    ((float*)Cout)[(size_t)r * N + c] = acc[s][t][j];
                }
            }
        }
    }
}

// ---------------------------------------------------------------------------
// Flash attention (causal), per block: one (b, h, 64-row Q tile).
// 128 threads = 4 waves; wave w owns q rows [qb+16w, +16).
// 64-key tiles staged in LDS (K row-major, V transposed):
//   per stage per wave: 8 QK^T WMMAs + 8 PV WMMAs.
// P goes through per-wave LDS buffer to convert C-layout -> A-layout.
// ---------------------------------------------------------------------------
__global__ __launch_bounds__(128) void mla_attn(const bf16_t* __restrict__ Q,
                                                const bf16_t* __restrict__ Kt,
                                                const bf16_t* __restrict__ Vt,
                                                bf16_t* __restrict__ O, int S) {
    __shared__ bf16_t Ks[64][72];      // [key][d]
    __shared__ bf16_t VT[64][72];      // [d][key]
    __shared__ bf16_t Ps[4][16][72];   // per-wave P: [qrow][key]

    const int b    = blockIdx.z;
    const int h    = blockIdx.y;
    const int qb   = blockIdx.x * 64;
    const int tid  = threadIdx.x;
    const int wave = tid >> 5;
    const int lane = tid & 31;
    const int m16  = lane & 15;
    const int hi   = lane >> 4;

    // Q A-fragments (16 rows x 64 d) : row per lane = qb + 16*wave + m16
    const int qrowA = qb + wave * 16 + m16;
    const size_t qoff = ((size_t)(b * S + qrowA)) * 1024 + h * 64;
    v16bf qf[2];
#pragma unroll
    for (int ks2 = 0; ks2 < 2; ++ks2)
        qf[ks2] = ld_frag(&Q[qoff + ks2 * 32 + hi * 8],
                          &Q[qoff + ks2 * 32 + 16 + hi * 8]);

    v8f o[4] = {v8f{}, v8f{}, v8f{}, v8f{}};
    float mrow[8], lrow[8];
#pragma unroll
    for (int j = 0; j < 8; ++j) { mrow[j] = -1e30f; lrow[j] = 0.0f; }

    const int ktiles = qb / 64 + 1;  // causal bound for this q tile
    for (int kt = 0; kt < ktiles; ++kt) {
        const int kbase = kt * 64;
        __syncthreads();
        // --- stage K tile (64 x 64) row-major ---
#pragma unroll
        for (int u = 0; u < 4; ++u) {
            int lin = tid + u * 128;
            int r = lin >> 3, seg = (lin & 7) * 8;
            *(bf16x8*)&Ks[r][seg] =
                *(const bf16x8*)&Kt[((size_t)(b * S + kbase + r)) * 1024 + h * 64 + seg];
        }
        // --- stage V tile transposed (d-major: VT[d][key]) ---
#pragma unroll
        for (int u = 0; u < 4; ++u) {
            int lin = tid + u * 128;
            int r = lin >> 3, seg = (lin & 7) * 8;
            bf16x8 v =
                *(const bf16x8*)&Vt[((size_t)(b * S + kbase + r)) * 1024 + h * 64 + seg];
#pragma unroll
            for (int i = 0; i < 8; ++i) VT[seg + i][r] = v[i];
        }
        __syncthreads();

        // --- scores S = Q * K^T  (16 q-rows x 64 keys, 8 WMMAs) ---
        v8f sf[4] = {v8f{}, v8f{}, v8f{}, v8f{}};
#pragma unroll
        for (int ks2 = 0; ks2 < 2; ++ks2) {
#pragma unroll
            for (int t = 0; t < 4; ++t) {
                v16bf bk = ld_frag(&Ks[t * 16 + m16][ks2 * 32 + hi * 8],
                                   &Ks[t * 16 + m16][ks2 * 32 + 16 + hi * 8]);
                sf[t] = wmma_bf16(qf[ks2], bk, sf[t]);
            }
        }

        // --- scale, causal mask, online softmax on C-fragment layout ---
#pragma unroll
        for (int j = 0; j < 8; ++j) {
            const int qrow = qb + wave * 16 + j + 8 * hi;
            float v[4];
#pragma unroll
            for (int t = 0; t < 4; ++t) {
                v[t] = sf[t][j] * 0.125f;
                if (kbase + t * 16 + m16 > qrow) v[t] = -1e30f;
            }
            float mx = fmaxf(fmaxf(v[0], v[1]), fmaxf(v[2], v[3]));
            mx = fmaxf(mx, __shfl_xor(mx, 1));
            mx = fmaxf(mx, __shfl_xor(mx, 2));
            mx = fmaxf(mx, __shfl_xor(mx, 4));
            mx = fmaxf(mx, __shfl_xor(mx, 8));   // 16-lane half-wave row max
            const float mn = fmaxf(mrow[j], mx);
            const float scale = __expf(mrow[j] - mn);
            float p[4], ps = 0.0f;
#pragma unroll
            for (int t = 0; t < 4; ++t) { p[t] = __expf(v[t] - mn); ps += p[t]; }
            ps += __shfl_xor(ps, 1);
            ps += __shfl_xor(ps, 2);
            ps += __shfl_xor(ps, 4);
            ps += __shfl_xor(ps, 8);             // row sum
            lrow[j] = lrow[j] * scale + ps;
            mrow[j] = mn;
            o[0][j] *= scale; o[1][j] *= scale;
            o[2][j] *= scale; o[3][j] *= scale;
            // C-layout -> LDS (row = j + 8*hi, cols t*16 + m16)
#pragma unroll
            for (int t = 0; t < 4; ++t)
                Ps[wave][j + 8 * hi][t * 16 + m16] = (bf16_t)p[t];
        }
        // ensure our wave's P stores have landed before A-layout reads
        asm volatile("s_wait_dscnt 0" ::: "memory");

        // --- O += P * V  (A-frags from Ps, B-frags from VT, 8 WMMAs) ---
        v16bf pf[2];
#pragma unroll
        for (int ks2 = 0; ks2 < 2; ++ks2)
            pf[ks2] = ld_frag(&Ps[wave][m16][ks2 * 32 + hi * 8],
                              &Ps[wave][m16][ks2 * 32 + 16 + hi * 8]);
#pragma unroll
        for (int t = 0; t < 4; ++t) {
#pragma unroll
            for (int ks2 = 0; ks2 < 2; ++ks2) {
                v16bf vfr = ld_frag(&VT[t * 16 + m16][ks2 * 32 + hi * 8],
                                    &VT[t * 16 + m16][ks2 * 32 + 16 + hi * 8]);
                o[t] = wmma_bf16(pf[ks2], vfr, o[t]);
            }
        }
    }

    // --- normalize and store (bf16, layout (B*S, H*64)) ---
#pragma unroll
    for (int j = 0; j < 8; ++j) {
        const float inv = 1.0f / lrow[j];
        const int qrow = qb + wave * 16 + j + 8 * hi;
        const size_t base = ((size_t)(b * S + qrow)) * 1024 + h * 64;
#pragma unroll
        for (int t = 0; t < 4; ++t)
            O[base + t * 16 + m16] = (bf16_t)(o[t][j] * inv);
    }
}

// ---------------------------------------------------------------------------
// Launcher
// ---------------------------------------------------------------------------
extern "C" void kernel_launch(void* const* d_in, const int* in_sizes, int n_in,
                              void* d_out, int out_size, void* d_ws, size_t ws_size,
                              hipStream_t stream) {
    (void)in_sizes; (void)n_in; (void)out_size; (void)ws_size;
    const float* X     = (const float*)d_in[0];
    const float* W_Q   = (const float*)d_in[1];
    const float* W_dn  = (const float*)d_in[2];
    const float* W_upK = (const float*)d_in[3];
    const float* W_upV = (const float*)d_in[4];
    const float* W_O   = (const float*)d_in[5];

    const int S  = 2048;
    const int BS = 2 * S;  // 4096 rows

    char* ws = (char*)d_ws;
    size_t off = 0;
    auto take = [&](size_t bytes) -> char* {
        char* p = ws + off;
        off += (bytes + 255) & ~(size_t)255;
        return p;
    };
    bf16_t* Xb  = (bf16_t*)take((size_t)BS * 1024 * 2);
    bf16_t* Wqb = (bf16_t*)take((size_t)1024 * 1024 * 2);
    bf16_t* Wdb = (bf16_t*)take((size_t)1024 * 256 * 2);
    bf16_t* Wkb = (bf16_t*)take((size_t)256 * 1024 * 2);
    bf16_t* Wvb = (bf16_t*)take((size_t)256 * 1024 * 2);
    bf16_t* Wob = (bf16_t*)take((size_t)1024 * 1024 * 2);
    bf16_t* Qb  = (bf16_t*)take((size_t)BS * 1024 * 2);
    bf16_t* Cb  = (bf16_t*)take((size_t)BS * 256 * 2);
    bf16_t* Kb  = (bf16_t*)take((size_t)BS * 1024 * 2);
    bf16_t* Vb  = (bf16_t*)take((size_t)BS * 1024 * 2);
    bf16_t* Ab  = (bf16_t*)take((size_t)BS * 1024 * 2);

    auto cvt = [&](const float* src, bf16_t* dst, int n) {
        f32_to_bf16<<<(n + 255) / 256, 256, 0, stream>>>(src, dst, n);
    };
    cvt(X, Xb, BS * 1024);
    cvt(W_Q, Wqb, 1024 * 1024);
    cvt(W_dn, Wdb, 1024 * 256);
    cvt(W_upK, Wkb, 256 * 1024);
    cvt(W_upV, Wvb, 256 * 1024);
    cvt(W_O, Wob, 1024 * 1024);

    const dim3 blk(256);
    // Q = X @ W_Q
    gemm_bf16<true><<<dim3(1024 / 128, BS / 128), blk, 0, stream>>>(Xb, Wqb, Qb, BS, 1024, 1024);
    // c_kv = X @ W_down
    gemm_bf16<true><<<dim3(256 / 128, BS / 128), blk, 0, stream>>>(Xb, Wdb, Cb, BS, 256, 1024);
    // RoPE(Q)
    rope_bf16<<<(BS * 512 + 255) / 256, 256, 0, stream>>>(Qb, BS * 512, S);
    // K = c_kv @ W_up_K ; V = c_kv @ W_up_V
    gemm_bf16<true><<<dim3(1024 / 128, BS / 128), blk, 0, stream>>>(Cb, Wkb, Kb, BS, 1024, 256);
    gemm_bf16<true><<<dim3(1024 / 128, BS / 128), blk, 0, stream>>>(Cb, Wvb, Vb, BS, 1024, 256);
    // RoPE(K)
    rope_bf16<<<(BS * 512 + 255) / 256, 256, 0, stream>>>(Kb, BS * 512, S);
    // flash attention: grid (q-tiles, heads, batch)
    mla_attn<<<dim3(S / 64, 16, 2), dim3(128), 0, stream>>>(Qb, Kb, Vb, Ab, S);
    // out = attn_out @ W_O  (fp32 store to d_out)
    gemm_bf16<false><<<dim3(1024 / 128, BS / 128), blk, 0, stream>>>(Ab, Wob, d_out, BS, 1024, 1024);
}
